// ScaledDotProductAttention_18313740550421
// MI455X (gfx1250) — matrix operand order repaired and checked by reading, not compiled, not run on hardware
//
#include <hip/hip_runtime.h>

typedef __attribute__((ext_vector_type(16))) __bf16 v16bf;
typedef __attribute__((ext_vector_type(8)))  __bf16 v8bf;
typedef __attribute__((ext_vector_type(8)))  float  v8f;
typedef __attribute__((ext_vector_type(4)))  float  v4f;

#define BH_   16
#define SEQ_  4096
#define HD_   64
#define TILE_ 64
#define NW_   4
#define KTS_  72   // padded row stride (bf16 elems) for transposed K: 144B rows, 16B aligned
#define MASK_NEG (-3.0e38f)

// row-major 64x64 fp32 -> bf16 LDS tile
__device__ __forceinline__ void stage_tile(__bf16* dst, const float* __restrict__ src,
                                           int tid, float scale) {
  const v4f* s4 = (const v4f*)src;
  #pragma unroll
  for (int i = 0; i < 8; ++i) {
    int fl = i * 128 + tid;
    v4f f = s4[fl];
    int b = fl * 4;
    dst[b + 0] = (__bf16)(f.x * scale);
    dst[b + 1] = (__bf16)(f.y * scale);
    dst[b + 2] = (__bf16)(f.z * scale);
    dst[b + 3] = (__bf16)(f.w * scale);
  }
}

// padded staging: dst[n * KTS_ + d] = K[n][d]  (so B-fragment K runs are contiguous)
__device__ __forceinline__ void stage_tile_T(__bf16* dst, const float* __restrict__ src,
                                             int tid) {
  const v4f* s4 = (const v4f*)src;
  #pragma unroll
  for (int i = 0; i < 8; ++i) {
    int fl = i * 128 + tid;
    v4f f = s4[fl];
    int n = fl >> 4;         // seq row within chunk (0..63)
    int d = (fl & 15) * 4;   // head dim (0..60)
    dst[n * KTS_ + d + 0] = (__bf16)f.x;
    dst[n * KTS_ + d + 1] = (__bf16)f.y;
    dst[n * KTS_ + d + 2] = (__bf16)f.z;
    dst[n * KTS_ + d + 3] = (__bf16)f.w;
  }
}

__device__ __forceinline__ v16bf cat8(v8bf lo, v8bf hi) {
  return __builtin_shufflevector(lo, hi, 0,1,2,3,4,5,6,7,8,9,10,11,12,13,14,15);
}

__global__ void __launch_bounds__(128)
fa_fwd_kernel(const float* __restrict__ qg, const float* __restrict__ kg,
              const float* __restrict__ vg, float* __restrict__ outg,
              float* __restrict__ attng)
{
  __shared__ __bf16 sQ[TILE_ * HD_];
  __shared__ __bf16 sKt[HD_ * KTS_];     // transposed K chunk
  __shared__ __bf16 sV[TILE_ * HD_];
  __shared__ __bf16 pbuf[NW_][16 * HD_];

  const int tid  = threadIdx.x;
  const int lane = tid & 31;
  const int w    = tid >> 5;       // wave id 0..3
  const int h    = lane >> 4;      // half-wave
  const int m    = lane & 15;      // col within C frag / row within A frag
  const int kb   = h * 8;          // A-frag K base per half-wave
  const int bx   = blockIdx.x;
  const int bh   = blockIdx.y;
  const int r0   = bx * TILE_;

  const size_t base = (size_t)bh * SEQ_ * HD_;

  // ---- zero-fill the strictly-masked upper-triangular attn region (NT stores) ----
  {
    const int colStart = (bx + 1) * TILE_;
    const int width = SEQ_ - colStart;
    if (width > 0) {
      const int w4  = width >> 2;
      const int tot = TILE_ * w4;
      const v4f z = {0.f, 0.f, 0.f, 0.f};
      for (int idx = tid; idx < tot; idx += 128) {
        int row = idx / w4;
        int c4  = idx - row * w4;
        v4f* p = (v4f*)(attng + (size_t)(bh * SEQ_ + r0 + row) * SEQ_ + colStart + c4 * 4);
        __builtin_nontemporal_store(z, p);
      }
    }
  }

  // ---- stage Q tile (pre-scaled by 1/temperature = 0.125) ----
  stage_tile(sQ, qg + base + (size_t)r0 * HD_, tid, 0.125f);
  __syncthreads();

  // resident Q A-fragments (two 16x32 slices of the wave's 16x64 Q rows)
  v16bf qa0, qa1;
  {
    const __bf16* qr = &sQ[(w * 16 + m) * HD_];
    qa0 = cat8(*(const v8bf*)(qr + kb),      *(const v8bf*)(qr + kb + 16));
    qa1 = cat8(*(const v8bf*)(qr + 32 + kb), *(const v8bf*)(qr + 32 + kb + 16));
  }

  float mrow[8], lrow[8];
  #pragma unroll
  for (int r = 0; r < 8; ++r) { mrow[r] = MASK_NEG; lrow[r] = 0.f; }

  const int nChunks = bx + 1;  // causal: only chunks 0..bx hold valid columns

  // ================= pass A: online softmax statistics =================
  for (int j = 0; j < nChunks; ++j) {
    __syncthreads();
    stage_tile_T(sKt, kg + base + (size_t)j * TILE_ * HD_, tid);
    if (j + 1 < nChunks)
      __builtin_prefetch(kg + base + (size_t)(j + 1) * TILE_ * HD_ + tid * 8, 0, 1);
    __syncthreads();

    // all four 16-column score fragments for this 64-col chunk
    v8f cblk[4];
    #pragma unroll
    for (int nb = 0; nb < 4; ++nb) {
      // B frag: lane = seq column; head-dim K in table order per lane
      const __bf16* k0 = &sKt[(nb * 16 + m) * KTS_ + kb];
      const __bf16* k1 = &sKt[(nb * 16 + m) * KTS_ + 32 + kb];
      v16bf b0 = cat8(*(const v8bf*)k0, *(const v8bf*)(k0 + 16));
      v16bf b1 = cat8(*(const v8bf*)k1, *(const v8bf*)(k1 + 16));
      v8f c = {};
      c = __builtin_amdgcn_wmma_f32_16x16x32_bf16(false, qa0, false, b0, (short)0, c, false, false);
      c = __builtin_amdgcn_wmma_f32_16x16x32_bf16(false, qa1, false, b1, (short)0, c, false, false);
      cblk[nb] = c;
    }
    if (j == bx) {
      #pragma unroll
      for (int nb = 0; nb < 4; ++nb) {
        const int col = j * TILE_ + nb * 16 + m;
        #pragma unroll
        for (int r = 0; r < 8; ++r)
          if (col > (r0 + w * 16 + kb + r)) cblk[nb][r] = MASK_NEG;
      }
    }
    // one cross-lane reduction per row per chunk
    #pragma unroll
    for (int r = 0; r < 8; ++r) {
      float mx = fmaxf(fmaxf(cblk[0][r], cblk[1][r]), fmaxf(cblk[2][r], cblk[3][r]));
      mx = fmaxf(mx, __shfl_xor(mx, 1, 16));
      mx = fmaxf(mx, __shfl_xor(mx, 2, 16));
      mx = fmaxf(mx, __shfl_xor(mx, 4, 16));
      mx = fmaxf(mx, __shfl_xor(mx, 8, 16));
      float mnew = fmaxf(mrow[r], mx);
      float es = __expf(cblk[0][r] - mnew) + __expf(cblk[1][r] - mnew)
               + __expf(cblk[2][r] - mnew) + __expf(cblk[3][r] - mnew);
      es += __shfl_xor(es, 1, 16);
      es += __shfl_xor(es, 2, 16);
      es += __shfl_xor(es, 4, 16);
      es += __shfl_xor(es, 8, 16);
      lrow[r] = lrow[r] * __expf(mrow[r] - mnew) + es;
      mrow[r] = mnew;
    }
  }

  float rinv[8];
  #pragma unroll
  for (int r = 0; r < 8; ++r) rinv[r] = 1.0f / lrow[r];

  // ================= pass B: write probs (NT) + accumulate out = P@V =================
  v8f acc[4];
  const v8f z8 = {};
  #pragma unroll
  for (int t = 0; t < 4; ++t) acc[t] = z8;

  for (int j = 0; j < nChunks; ++j) {
    __syncthreads();
    stage_tile_T(sKt, kg + base + (size_t)j * TILE_ * HD_, tid);
    stage_tile  (sV,  vg + base + (size_t)j * TILE_ * HD_, tid, 1.0f);
    if (j + 1 < nChunks)
      __builtin_prefetch(vg + base + (size_t)(j + 1) * TILE_ * HD_ + tid * 8, 0, 1);
    __syncthreads();
    const bool diag = (j == bx);
    #pragma unroll
    for (int nb = 0; nb < 4; ++nb) {
      const __bf16* k0 = &sKt[(nb * 16 + m) * KTS_ + kb];
      const __bf16* k1 = &sKt[(nb * 16 + m) * KTS_ + 32 + kb];
      v16bf b0 = cat8(*(const v8bf*)k0, *(const v8bf*)(k0 + 16));
      v16bf b1 = cat8(*(const v8bf*)k1, *(const v8bf*)(k1 + 16));
      v8f c = {};
      c = __builtin_amdgcn_wmma_f32_16x16x32_bf16(false, qa0, false, b0, (short)0, c, false, false);
      c = __builtin_amdgcn_wmma_f32_16x16x32_bf16(false, qa1, false, b1, (short)0, c, false, false);
      const int col = j * TILE_ + nb * 16 + m;
      #pragma unroll
      for (int r = 0; r < 8; ++r) {
        const int rowl = kb + r;                 // row in wave tile (0..15)
        const int rowg = r0 + w * 16 + rowl;
        float s = c[r];
        if (diag && col > rowg) s = MASK_NEG;
        float p = __expf(s - mrow[r]) * rinv[r];
        __builtin_nontemporal_store(p, attng + (size_t)(bh * SEQ_ + rowg) * SEQ_ + col);
        pbuf[w][rowl * HD_ + nb * 16 + m] = (__bf16)p;
      }
    }
    // per-wave LDS round trip: DS stores above must land before fragment reloads
    asm volatile("s_wait_dscnt 0" ::: "memory");
    v16bf pa0, pa1;
    {
      const __bf16* pr = &pbuf[w][m * HD_];
      pa0 = cat8(*(const v8bf*)(pr + kb),      *(const v8bf*)(pr + kb + 16));
      pa1 = cat8(*(const v8bf*)(pr + 32 + kb), *(const v8bf*)(pr + 32 + kb + 16));
    }
    #pragma unroll
    for (int nb2 = 0; nb2 < 4; ++nb2) {
      v16bf bv0, bv1;
      #pragma unroll
      for (int s = 0; s < 16; ++s) {
        const int kk = kb + (s & 7) + ((s >> 3) << 4);
        bv0[s] = sV[kk * HD_ + nb2 * 16 + m];
        bv1[s] = sV[(32 + kk) * HD_ + nb2 * 16 + m];
      }
      acc[nb2] = __builtin_amdgcn_wmma_f32_16x16x32_bf16(false, pa0, false, bv0, (short)0, acc[nb2], false, false);
      acc[nb2] = __builtin_amdgcn_wmma_f32_16x16x32_bf16(false, pa1, false, bv1, (short)0, acc[nb2], false, false);
    }
  }

  // ---- write out (16 MB total, coalesced across lanes) ----
  #pragma unroll
  for (int nb2 = 0; nb2 < 4; ++nb2) {
    #pragma unroll
    for (int r = 0; r < 8; ++r) {
      outg[(size_t)(bh * SEQ_ + r0 + w * 16 + kb + r) * HD_ + nb2 * 16 + m] = acc[nb2][r];
    }
  }
}

extern "C" void kernel_launch(void* const* d_in, const int* in_sizes, int n_in,
                              void* d_out, int out_size, void* d_ws, size_t ws_size,
                              hipStream_t stream) {
  (void)in_sizes; (void)n_in; (void)out_size; (void)d_ws; (void)ws_size;
  const float* q = (const float*)d_in[0];
  const float* k = (const float*)d_in[1];
  const float* v = (const float*)d_in[2];
  // d_in[3] is the causal mask; the kernel applies causality analytically.
  float* out  = (float*)d_out;
  float* attn = out + (size_t)BH_ * SEQ_ * HD_;
  dim3 grid(SEQ_ / TILE_, BH_);
  fa_fwd_kernel<<<grid, dim3(128), 0, stream>>>(q, k, v, out, attn);
}
